// FeatureEmbed_3874060501841
// MI455X (gfx1250) — matrix operand order repaired and checked by reading, not compiled, not run on hardware
//
#include <hip/hip_runtime.h>
#include <hip/hip_bf16.h>

// ---------------------------------------------------------------------------
// FeatureEmbed: out[cell, c, w] = sum_g x[cell,g] * W[g, w*64+c] * mask[g,(w*64+c)/192]
//                                 + b[w*64+c]
// GEMM M=4096, K=4000, N=12288 with fused binary mask and permuted writeout.
// f16 WMMA (v_wmma_f32_16x16x32_f16) with f32 accumulation; double-buffered LDS.
// Round 2: batch all 12 ds_load_b128 fragment loads before the 8 WMMAs so the
// compiler emits one DS wait per K-step instead of s_wait_dscnt 0 every 2 WMMAs.
// ---------------------------------------------------------------------------

typedef __attribute__((ext_vector_type(16))) _Float16 v16h;
typedef __attribute__((ext_vector_type(8)))  _Float16 v8h;
typedef __attribute__((ext_vector_type(4)))  _Float16 v4h;
typedef __attribute__((ext_vector_type(8)))  float    v8f;
typedef __attribute__((ext_vector_type(4)))  float    v4f;

#define GENES   4000
#define PATCHES 64
#define EMBED   192
#define CELLS   4096
#define NCOL    (PATCHES * EMBED)   // 12288
#define KT      32
#define MT      128
#define NT      128
#define KSTEPS  (GENES / KT)        // 125
#define LROW    40                  // halfs per LDS row: 32 data + 8 pad (80B -> conflict-free b128)

union AFrag { v16h v; v8h h[2]; };

__global__ __launch_bounds__(256)
void feature_embed_wmma(const float* __restrict__ x,
                        const float* __restrict__ mask,
                        const float* __restrict__ W,
                        const float* __restrict__ bias,
                        float* __restrict__ out) {
    __shared__ _Float16 lA[2][MT * LROW];   // A tile, [m][k], f16
    __shared__ _Float16 lB[2][NT * LROW];   // B tile, transposed [n][k], f16 (mask fused)

    const int t      = threadIdx.x;
    const int lane   = t & 31;
    const int wid    = t >> 5;
    const int wave_m = wid >> 2;   // 0..1  -> 64 rows each
    const int wave_n = wid & 3;    // 0..3  -> 32 cols each
    const int t_lo   = lane & 15;
    const int t_hi   = lane >> 4;

    const int n0 = blockIdx.x * NT;   // column tile (W / output)
    const int c0 = blockIdx.y * MT;   // cell tile   (x rows)

    // Staging coordinates (global f32 -> LDS f16), float4-vectorized global reads.
    const int a_k4   = (t & 7)  * 4;   // k offset within tile: 0..28
    const int a_row0 = (t >> 3) * 4;   // m offset within tile: 0..124
    const int b_n4   = (t & 31) * 4;   // n offset within tile: 0..124
    const int b_g0   = (t >> 5) * 4;   // k offset within tile: 0..28

    v4f   pa[4], pb[4];
    float pm[4][4];

    auto prefetch = [&](int kt) {
        const int k0 = kt * KT;
#pragma unroll
        for (int r = 0; r < 4; ++r)
            pa[r] = *(const v4f*)(x + (size_t)(c0 + a_row0 + r) * GENES + k0 + a_k4);
#pragma unroll
        for (int r = 0; r < 4; ++r) {
            const int g = k0 + b_g0 + r;
            pb[r] = *(const v4f*)(W + (size_t)g * NCOL + n0 + b_n4);
#pragma unroll
            for (int e = 0; e < 4; ++e)
                pm[r][e] = mask[(size_t)g * PATCHES + (unsigned)(n0 + b_n4 + e) / EMBED];
        }
    };

    auto stage = [&](int buf) {
#pragma unroll
        for (int r = 0; r < 4; ++r) {
            v4h av = { (_Float16)pa[r][0], (_Float16)pa[r][1],
                       (_Float16)pa[r][2], (_Float16)pa[r][3] };
            *(v4h*)&lA[buf][(a_row0 + r) * LROW + a_k4] = av;   // 8B aligned ds_store
        }
#pragma unroll
        for (int r = 0; r < 4; ++r)
#pragma unroll
            for (int e = 0; e < 4; ++e)
                lB[buf][(b_n4 + e) * LROW + b_g0 + r] =
                    (_Float16)(pb[r][e] * pm[r][e]);            // transpose into [n][k]
    };

    v8f acc[4][2] = {};

    auto compute = [&](int buf) {
        // Issue ALL fragment loads first (12x ds_load_b128), then consume:
        // one DS wait per K-step, 8 back-to-back WMMAs (distinct accumulators).
        v8h a0[4], a1[4], b0[2], b1[2];
#pragma unroll
        for (int j = 0; j < 2; ++j) {
            // B frag (32x16): lane n = wave_n*32 + j*16 + t_lo, 16 contiguous k at t_hi*16
            const _Float16* p = &lB[buf][(wave_n * 32 + j * 16 + t_lo) * LROW + t_hi * 16];
            b0[j] = *(const v8h*)p;              // 16B aligned
            b1[j] = *(const v8h*)(p + 8);
        }
#pragma unroll
        for (int i = 0; i < 4; ++i) {
            // A frag (16x32): lane m = wave_m*64 + i*16 + t_lo, 16 contiguous k at t_hi*16
            const _Float16* p = &lA[buf][(wave_m * 64 + i * 16 + t_lo) * LROW + t_hi * 16];
            a0[i] = *(const v8h*)p;
            a1[i] = *(const v8h*)(p + 8);
        }
#pragma unroll
        for (int i = 0; i < 4; ++i) {
            AFrag af; af.h[0] = a0[i]; af.h[1] = a1[i];
#pragma unroll
            for (int j = 0; j < 2; ++j) {
                AFrag bf; bf.h[0] = b0[j]; bf.h[1] = b1[j];
                acc[i][j] = __builtin_amdgcn_wmma_f32_16x16x32_f16(
                    false, af.v, false, bf.v, (short)0, acc[i][j], false, false);
            }
        }
    };

    prefetch(0);
    stage(0);
    __syncthreads();

    for (int kt = 0; kt < KSTEPS; ++kt) {
        const int buf = kt & 1;
        if (kt + 1 < KSTEPS) prefetch(kt + 1);   // hide global latency behind WMMAs
        compute(buf);
        if (kt + 1 < KSTEPS) stage(buf ^ 1);
        __syncthreads();                          // one barrier/iter: double-buffer safe
    }

    // Epilogue: bias + permuted store. C/D layout: lane holds n = t_lo, m = r + 8*t_hi.
#pragma unroll
    for (int i = 0; i < 4; ++i) {
#pragma unroll
        for (int j = 0; j < 2; ++j) {
            const int   jg = n0 + wave_n * 32 + j * 16 + t_lo;   // GEMM column
            const float bv = bias[jg];
            const int   cc = jg & 63;    // patch slot in output (j % 64)
            const int   ww = jg >> 6;    // embed slot in output (j / 64)
#pragma unroll
            for (int r = 0; r < 8; ++r) {
                const int cell = c0 + wave_m * 64 + i * 16 + t_hi * 8 + r;
                out[(size_t)cell * NCOL + cc * EMBED + ww] = acc[i][j][r] + bv;
            }
        }
    }
}

extern "C" void kernel_launch(void* const* d_in, const int* in_sizes, int n_in,
                              void* d_out, int out_size, void* d_ws, size_t ws_size,
                              hipStream_t stream) {
    (void)in_sizes; (void)n_in; (void)d_ws; (void)ws_size; (void)out_size;
    const float* x    = (const float*)d_in[0];   // [4096, 4000]
    const float* mask = (const float*)d_in[1];   // [4000, 64]
    const float* W    = (const float*)d_in[2];   // [4000, 12288]
    const float* b    = (const float*)d_in[3];   // [12288]
    float* out = (float*)d_out;                  // [4096, 64, 192]

    dim3 grid(NCOL / NT, CELLS / MT);  // (96, 32): x-panel L2-resident across N sweep
    dim3 block(256);
    feature_embed_wmma<<<grid, block, 0, stream>>>(x, mask, W, b, out);
}
